// Model_54090818126165
// MI455X (gfx1250) — matrix operand order repaired and checked by reading, not compiled, not run on hardware
//
#include <hip/hip_runtime.h>
#include <hip/hip_bf16.h>
#include <math.h>

// ---------------------------------------------------------------------------
// MI455X (gfx1250, CDNA5) implementation.
// Live subgraph only (c3/c4/c5 + nl2..nl4 are dead w.r.t. outputs).
// Convolutions = implicit GEMM on v_wmma_f32_16x16x32_bf16, f32 accumulate.
// Block tile M=32 x N=128, 8 waves, each wave 16x32 (2 WMMA sharing A frag).
// ---------------------------------------------------------------------------

typedef __bf16 bf16_t;
typedef __bf16 v16bf  __attribute__((ext_vector_type(16)));
typedef __bf16 bf16x4 __attribute__((ext_vector_type(4)));
typedef float  v8f    __attribute__((ext_vector_type(8)));

#define EPS_BN  1e-5f
#define EPS_COS 1e-8f

// ----------------------------- weight packing ------------------------------
__global__ void pack_wgt_kernel(const float* __restrict__ w, bf16_t* __restrict__ A,
                                int cout, int cin_src, int cin_eff, int ks2,
                                int Kpad, int Mpad, int fold)
{
    int idx = blockIdx.x * blockDim.x + threadIdx.x;
    int total = Mpad * Kpad;
    if (idx >= total) return;
    int row = idx / Kpad;
    int col = idx - row * Kpad;
    float v = 0.f;
    int K = cin_eff * ks2;
    if (row < cout && col < K) {
        int c = col / ks2;
        int t = col - c * ks2;
        v = w[(row * cin_src + c) * ks2 + t];
        if (fold) v += w[(row * cin_src + c + cin_eff) * ks2 + t];
    }
    A[idx] = (bf16_t)v;
}

__global__ void fuse_bn_kernel(const float* __restrict__ b,
                               const float* __restrict__ gamma, const float* __restrict__ beta,
                               const float* __restrict__ mean,  const float* __restrict__ var,
                               float* __restrict__ scale, float* __restrict__ shift,
                               int C, int has_bn)
{
    int c = blockIdx.x * blockDim.x + threadIdx.x;
    if (c >= C) return;
    if (has_bn) {
        float s = gamma[c] * rsqrtf(var[c] + EPS_BN);
        scale[c] = s;
        shift[c] = beta[c] - mean[c] * s + b[c] * s;
    } else {
        scale[c] = 1.f;
        shift[c] = b[c];
    }
}

// --------------------------- implicit-GEMM conv ----------------------------
#define AS 40    // A tile LDS row stride (bf16): 80B, 8B-aligned, conflict-free
#define BS 132   // B tile LDS row stride (bf16): 264B, 8B-aligned, conflict-free

__global__ __launch_bounds__(256)
void conv_gemm_wmma(const float* __restrict__ in, float* __restrict__ out,
                    const bf16_t* __restrict__ A,
                    const float* __restrict__ scale, const float* __restrict__ shift,
                    const float* __restrict__ residual,
                    int NB, int Cin, int H, int W, int Cout, int ks, int act, int Kpad)
{
    __shared__ bf16_t Asm[32 * AS];
    __shared__ bf16_t Bsm[32 * BS];

    const int HWp  = H * W;
    const int NTOT = NB * HWp;
    const int ks2  = ks * ks;
    const int KK   = Cin * ks2;
    const int pad  = ks >> 1;

    const int n0   = blockIdx.x * 128;
    const int m0   = blockIdx.y * 32;
    const int tid  = threadIdx.x;
    const int lane = tid & 31;
    const int wave = tid >> 5;
    const int wm   = wave >> 2;    // 0..1: M sub-tile
    const int wn   = wave & 3;     // 0..3: N sub-tile (32 cols each)
    const int l15  = lane & 15;
    const int hi   = lane >> 4;

    // ---- A staging geometry: one 8B (4x bf16) load per thread ----
    const int ar = tid >> 3;          // 0..31 (M row)
    const int ac = (tid & 7) * 4;     // 0..28 (K col, quad)

    // ---- B staging geometry (invariant over the K loop) ----
    // thread -> one 4-wide N quad (q) x four K rows (rb + 8*i).
    const int q  = lane;              // reuse: q = tid & 31
    const int rb = wave;              // k-row base, rows rb + 8*i
    const int nq = n0 + q * 4;        // W,HW multiples of 4 -> quad stays in one image row
    const bool qvalid = nq < NTOT;
    int y_ = 0, x_ = 0;
    size_t base_in = 0;
    if (qvalid) {
        int b_  = nq / HWp;
        int rem = nq - b_ * HWp;
        y_ = rem / W;
        x_ = rem - y_ * W;
        base_in = (size_t)b_ * Cin * HWp + (size_t)y_ * W + x_;
    }
    // incremental kg -> (ci = kg/ks2, t = kg%ks2); step per K iter: +32
    int ci_[4], t_[4];
    #pragma unroll
    for (int i = 0; i < 4; ++i) {
        int kg = rb + 8 * i;
        ci_[i] = kg / ks2;
        t_[i]  = kg - ci_[i] * ks2;
    }
    const int dci = 32 / ks2;
    const int dt  = 32 - dci * ks2;    // 32 % ks2

    v8f acc[2];
    #pragma unroll
    for (int s = 0; s < 2; ++s)
        acc[s] = (v8f){0.f,0.f,0.f,0.f,0.f,0.f,0.f,0.f};

    for (int k0 = 0; k0 < Kpad; k0 += 32) {
        // ---- stage A tile: vectorized 4x bf16 per thread ----
        *(bf16x4*)(&Asm[ar * AS + ac]) =
            *(const bf16x4*)(A + (size_t)(m0 + ar) * Kpad + k0 + ac);
        if (k0 + 32 < Kpad)                         // -> global_prefetch_b8
            __builtin_prefetch(A + (size_t)m0 * Kpad + k0 + 32, 0, 1);

        // ---- stage B tile: 4 K-rows x 4 N-cols per thread, im2col gather ----
        #pragma unroll
        for (int i = 0; i < 4; ++i) {
            int r  = rb + 8 * i;
            int kg = k0 + r;
            int ci = ci_[i], t = t_[i];
            float v0 = 0.f, v1 = 0.f, v2 = 0.f, v3 = 0.f;
            if (qvalid && kg < KK) {
                int ty = (ks == 3) ? ((t >= 6) ? 2 : ((t >= 3) ? 1 : 0)) : 0;
                int dy = ty - pad;
                int dx = (t - ty * ks) - pad;
                int iy = y_ + dy;
                if (iy >= 0 && iy < H) {
                    const float* p = in + base_in + (size_t)ci * HWp + dy * W + dx;
                    int ix0 = x_ + dx;
                    v0 = (ix0 >= 0)     ? p[0] : 0.f;   // only j=0 can underflow
                    v1 = p[1];                          // always in-row
                    v2 = p[2];
                    v3 = (ix0 + 3 < W)  ? p[3] : 0.f;   // only j=3 can overflow
                }
            }
            bf16x4 pk = { (bf16_t)v0, (bf16_t)v1, (bf16_t)v2, (bf16_t)v3 };
            *(bf16x4*)(&Bsm[r * BS + q * 4]) = pk;
            // advance (ci, t) by +32 for the next K step (no division)
            int tn = t + dt;
            int cr = (tn >= ks2) ? 1 : 0;
            ci_[i] = ci + dci + cr;
            t_[i]  = tn - (cr ? ks2 : 0);
        }
        __syncthreads();

        // ---- fragments (ISA 7.12.2): A 16x32 per wave, B lane=N / in-lane=K
        v16bf af;
        {
            int arow = wm * 16 + l15;
            #pragma unroll
            for (int vr = 0; vr < 8; ++vr) {
                int kb = ((vr >= 4) ? 16 : 0) + hi * 8 + (vr & 3) * 2;
                af[2 * vr]     = Asm[arow * AS + kb];
                af[2 * vr + 1] = Asm[arow * AS + kb + 1];
            }
        }
        #pragma unroll
        for (int s = 0; s < 2; ++s) {
            v16bf bfg;
            int bcol = wn * 32 + s * 16;
            #pragma unroll
            for (int vr = 0; vr < 8; ++vr) {
                int kb = ((vr >= 4) ? 16 : 0) + hi * 8 + (vr & 3) * 2;
                bfg[2 * vr]     = Bsm[kb * BS + bcol + l15];
                bfg[2 * vr + 1] = Bsm[(kb + 1) * BS + bcol + l15];
            }
            acc[s] = __builtin_amdgcn_wmma_f32_16x16x32_bf16(false, af, false, bfg,
                                                             (short)0, acc[s], false, false);
        }
        __syncthreads();
    }

    // ---- fused epilogue: BN scale/shift (+bias), residual, activation ----
    #pragma unroll
    for (int s = 0; s < 2; ++s) {
        int ncol = n0 + wn * 32 + s * 16 + l15;
        if (ncol < NTOT) {
            int b   = ncol / HWp;
            int rem = ncol - b * HWp;
            #pragma unroll
            for (int vr = 0; vr < 8; ++vr) {
                int m = m0 + wm * 16 + vr + hi * 8;
                if (m < Cout) {
                    float r = acc[s][vr] * scale[m] + shift[m];
                    size_t o = (size_t)(b * Cout + m) * HWp + rem;
                    if (residual) r += residual[o];
                    if (act == 1)      r = fmaxf(r, 0.f);
                    else if (act == 2) r = 1.f / (1.f + __expf(-r));
                    out[o] = r;
                }
            }
        }
    }
}

// ------------------------------ max-pool 2x2 -------------------------------
__global__ void maxpool2x2_kernel(const float* __restrict__ in, float* __restrict__ out,
                                  int BC, int H, int W)
{
    int Ho = H >> 1, Wo = W >> 1;
    int idx = blockIdx.x * blockDim.x + threadIdx.x;
    int tot = BC * Ho * Wo;
    if (idx >= tot) return;
    int p  = idx % (Ho * Wo);
    int bc = idx / (Ho * Wo);
    int y = p / Wo, x = p - y * Wo;
    const float* src = in + ((size_t)bc * H + 2 * y) * W + 2 * x;
    out[(size_t)bc * Ho * Wo + p] =
        fmaxf(fmaxf(src[0], src[1]), fmaxf(src[W], src[W + 1]));
}

// ------------------ WeightAverage cosine-softmax attention -----------------
__global__ void wavg_attn_kernel(const float* __restrict__ Th, const float* __restrict__ Ph,
                                 const float* __restrict__ G,  float* __restrict__ wa,
                                 int NB, int C, int H, int W)
{
    int HWp = H * W;
    int idx = blockIdx.x * blockDim.x + threadIdx.x;
    if (idx >= NB * HWp) return;
    int b   = idx / HWp;
    int rem = idx - b * HWp;
    int y = rem / W, x = rem - y * W;

    int off[9];
    #pragma unroll
    for (int j = 0; j < 9; ++j) {
        int dy = j / 3 - 1, dx = j % 3 - 1;
        int iy = min(max(y + dy, 0), H - 1);
        int ix = min(max(x + dx, 0), W - 1);
        off[j] = iy * W + ix;
    }
    float dot[9], np2[9];
    #pragma unroll
    for (int j = 0; j < 9; ++j) { dot[j] = 0.f; np2[j] = 0.f; }
    float nth2 = 0.f;
    size_t base = (size_t)b * C * HWp;
    for (int c = 0; c < C; ++c) {
        const float* phc = Ph + base + (size_t)c * HWp;
        float t = Th[base + (size_t)c * HWp + rem];
        nth2 += t * t;
        #pragma unroll
        for (int j = 0; j < 9; ++j) {
            float p = phc[off[j]];
            dot[j] += p * t;
            np2[j] += p * p;
        }
    }
    float nth = fmaxf(sqrtf(nth2), EPS_COS);
    float cv[9]; float mx = -1e30f;
    #pragma unroll
    for (int j = 0; j < 9; ++j) {
        cv[j] = dot[j] / (fmaxf(sqrtf(np2[j]), EPS_COS) * nth);
        mx = fmaxf(mx, cv[j]);
    }
    float s = 0.f;
    #pragma unroll
    for (int j = 0; j < 9; ++j) { cv[j] = __expf(cv[j] - mx); s += cv[j]; }
    float inv = 1.f / s;
    for (int c = 0; c < C; ++c) {
        const float* gc = G + base + (size_t)c * HWp;
        float a = 0.f;
        #pragma unroll
        for (int j = 0; j < 9; ++j) a += cv[j] * gc[off[j]];
        wa[base + (size_t)c * HWp + rem] = a * inv;
    }
}

// --------------------------- amp * dmp broadcast ---------------------------
__global__ void mul_bcast_kernel(float* __restrict__ d, const float* __restrict__ a,
                                 int NB, int C, int HWp)
{
    int idx = blockIdx.x * blockDim.x + threadIdx.x;
    int tot = NB * C * HWp;
    if (idx >= tot) return;
    int b = idx / (C * HWp);
    int p = idx % HWp;
    d[idx] *= a[(size_t)b * HWp + p];
}

// ------------------------------- host driver -------------------------------
struct LSpec { int wi, bi; int bn; int cout, cin_src, cin_eff, ks; int fold; };

static inline size_t al256(size_t v) { return (v + 255) & ~(size_t)255; }

extern "C" void kernel_launch(void* const* d_in, const int* in_sizes, int n_in,
                              void* d_out, int out_size, void* d_ws, size_t ws_size,
                              hipStream_t stream)
{
    (void)in_sizes; (void)n_in; (void)out_size; (void)ws_size;
    const float* x = (const float*)d_in[0];
    char* ws = (char*)d_ws;

    const int NB = 2;
    const int H1 = 384, W1 = 384, H2 = 192, W2 = 192;
    const size_t HW2 = (size_t)H2 * W2;
    const size_t SB = (size_t)NB * 64  * H1 * W1 * 4;  // 64ch @384^2
    const size_t S1 = (size_t)NB * 64  * HW2 * 4;      // 64ch @192^2
    const size_t S2 = (size_t)NB * 128 * HW2 * 4;      // 128ch@192^2
    const size_t S3 = (size_t)NB * 32  * HW2 * 4;      // 32ch @192^2

    // liveness-based reuse inside a 2*SB (~151MB) activation pool
    float* buf_a  = (float*)(ws + 0);
    float* buf_b  = (float*)(ws + SB);
    float* pool_o = (float*)(ws + 0);                  // reuse buf_a
    float* a21    = (float*)(ws + S1);
    float* c2     = (float*)(ws + S1 + S2);
    float* th     = (float*)(ws + S1 + 2 * S2);
    float* ph     = (float*)(ws + 2 * S1 + 2 * S2);
    float* gg     = (float*)(ws + 3 * S1 + 2 * S2);
    float* wab    = (float*)(ws + 0);                  // reuse pool_o
    float* c2n    = (float*)(ws + S1);                 // reuse a21
    float* amp5   = (float*)(ws + S1 + S2);            // reuse c2
    float* amp6   = (float*)(ws + 2 * S1 + S2);
    float* amp7   = (float*)(ws + 3 * S1 + S2);
    float* dmp5   = (float*)(ws + 0);                  // reuse wab
    float* dmp6   = (float*)(ws + 3 * S1 + S2 + S3);
    float* dmp7   = (float*)(ws + 4 * S1 + S2 + S3);

    size_t woff = 2 * SB;                               // weights region

    static const LSpec Ls[16] = {
        {  1,  2, 1,  64,   3,   3, 3, 0},  // 0  c1_1
        {  7,  8, 1,  64,  64,  64, 3, 0},  // 1  c1_2
        { 13, 14, 1, 128,  64,  64, 3, 0},  // 2  c2_1
        { 19, 20, 1, 128, 128, 128, 3, 0},  // 3  c2_2
        { 79, 80, 0,  64, 128, 128, 1, 0},  // 4  nl1.theta
        { 81, 82, 0,  64, 128, 128, 1, 0},  // 5  nl1.phi
        { 83, 84, 0,  64, 128, 128, 1, 0},  // 6  nl1.g
        { 85, 86, 0, 128,  64,  64, 1, 0},  // 7  nl1.back
        {111,112, 1,  64, 256, 128, 1, 1},  // 8  amp.c5 (cat fold)
        {117,118, 1,  64,  64,  64, 3, 0},  // 9  amp.c6
        {123,124, 1,  32,  64,  64, 3, 0},  // 10 amp.c7
        {129,130, 1,  64, 256, 128, 1, 1},  // 11 dmp.c5 (cat fold)
        {135,136, 1,  64,  64,  64, 3, 0},  // 12 dmp.c6
        {141,142, 1,  32,  64,  64, 3, 0},  // 13 dmp.c7
        {147,148, 1,   1,  32,  32, 1, 0},  // 14 att (sigmoid)
        {153,154, 0,   1,  32,  32, 1, 0},  // 15 out
    };

    bf16_t* Abf[16]; float* Scl[16]; float* Shf[16]; int Kp[16], Mp[16];
    for (int i = 0; i < 16; ++i) {
        const LSpec& L = Ls[i];
        int ks2  = L.ks * L.ks;
        int K    = L.cin_eff * ks2;
        int Kpad = (K + 31) & ~31;
        int Mpad = (L.cout + 31) & ~31;
        Kp[i] = Kpad; Mp[i] = Mpad;
        Abf[i] = (bf16_t*)(ws + woff); woff = al256(woff + (size_t)Mpad * Kpad * 2);
        Scl[i] = (float*)(ws + woff);  woff = al256(woff + (size_t)L.cout * 4);
        Shf[i] = (float*)(ws + woff);  woff = al256(woff + (size_t)L.cout * 4);
        int tot = Mpad * Kpad;
        pack_wgt_kernel<<<(tot + 255) / 256, 256, 0, stream>>>(
            (const float*)d_in[L.wi], Abf[i], L.cout, L.cin_src, L.cin_eff,
            ks2, Kpad, Mpad, L.fold);
        const float* bb = (const float*)d_in[L.bi];
        const float *g_ = nullptr, *be = nullptr, *me = nullptr, *va = nullptr;
        if (L.bn) {
            g_ = (const float*)d_in[L.bi + 1]; be = (const float*)d_in[L.bi + 2];
            me = (const float*)d_in[L.bi + 3]; va = (const float*)d_in[L.bi + 4];
        }
        fuse_bn_kernel<<<(L.cout + 255) / 256, 256, 0, stream>>>(
            bb, g_, be, me, va, Scl[i], Shf[i], L.cout, L.bn);
    }

    auto conv = [&](const float* in, float* out, int li, int Cin, int Hc, int Wc,
                    int Cout, int ks, int act, const float* resid) {
        int ntot = NB * Hc * Wc;
        dim3 g((ntot + 127) / 128, Mp[li] / 32);
        conv_gemm_wmma<<<g, 256, 0, stream>>>(in, out, Abf[li], Scl[li], Shf[li],
                                              resid, NB, Cin, Hc, Wc, Cout, ks,
                                              act, Kp[li]);
    };

    // -------- live network --------
    conv(x,      buf_a, 0,   3, H1, W1,  64, 3, 1, nullptr);     // c1_1
    conv(buf_a,  buf_b, 1,  64, H1, W1,  64, 3, 1, nullptr);     // c1_2
    {   int bc = NB * 64; int tot = bc * (H1/2) * (W1/2);
        maxpool2x2_kernel<<<(tot + 255)/256, 256, 0, stream>>>(buf_b, pool_o, bc, H1, W1); }
    conv(pool_o, a21,   2,  64, H2, W2, 128, 3, 1, nullptr);     // c2_1
    conv(a21,    c2,    3, 128, H2, W2, 128, 3, 1, nullptr);     // c2_2
    conv(c2,     th,    4, 128, H2, W2,  64, 1, 0, nullptr);     // theta
    conv(c2,     ph,    5, 128, H2, W2,  64, 1, 0, nullptr);     // phi
    conv(c2,     gg,    6, 128, H2, W2,  64, 1, 0, nullptr);     // g
    {   int tot = NB * (int)HW2;
        wavg_attn_kernel<<<(tot + 255)/256, 256, 0, stream>>>(th, ph, gg, wab, NB, 64, H2, W2); }
    conv(wab,    c2n,   7,  64, H2, W2, 128, 1, 0, c2);          // back + residual
    conv(c2n,    amp5,  8, 128, H2, W2,  64, 1, 1, nullptr);     // amp.c5 (folded cat)
    conv(amp5,   amp6,  9,  64, H2, W2,  64, 3, 1, nullptr);     // amp.c6
    conv(amp6,   amp7, 10,  64, H2, W2,  32, 3, 1, nullptr);     // amp.c7

    float* out_f   = (float*)d_out;
    float* amp_out = out_f + NB * HW2;                           // outputs: [dmp | amp]
    conv(amp7,   amp_out, 14, 32, H2, W2, 1, 1, 2, nullptr);     // att -> sigmoid

    conv(c2n,    dmp5, 11, 128, H2, W2,  64, 1, 1, nullptr);     // dmp.c5
    conv(dmp5,   dmp6, 12,  64, H2, W2,  64, 3, 1, nullptr);     // dmp.c6
    conv(dmp6,   dmp7, 13,  64, H2, W2,  32, 3, 1, nullptr);     // dmp.c7
    {   int tot = NB * 32 * (int)HW2;
        mul_bcast_kernel<<<(tot + 255)/256, 256, 0, stream>>>(dmp7, amp_out, NB, 32, (int)HW2); }
    conv(dmp7,   out_f, 15, 32, H2, W2, 1, 1, 0, nullptr);       // out conv -> dmp
}
